// BoxAwareAttention_32220844654735
// MI455X (gfx1250) — matrix operand order repaired and checked by reading, not compiled, not run on hardware
//
#include <hip/hip_runtime.h>
#include <hip/hip_bf16.h>
#include <stdint.h>

// ---------------------------------------------------------------------------
// Types for the gfx1250 WMMA builtins
// ---------------------------------------------------------------------------
typedef __attribute__((ext_vector_type(16))) __bf16    v16bf;
typedef __attribute__((ext_vector_type(8)))  float     v8f;
typedef __attribute__((ext_vector_type(4)))  unsigned  v4u;

// Problem constants (B=2, L=2048, C=1024, H=16, D=64)
#define BATCH 2
#define SEQ   2048
#define CDIM  1024
#define HEADS 16
#define HDIM  64
#define MROWS (BATCH * SEQ)          // 4096
#define QKVN  (3 * CDIM)             // 3072

// ---------------------------------------------------------------------------
// bf16 helpers (bit-level, RNE)
// ---------------------------------------------------------------------------
__device__ __forceinline__ float bf2f(__bf16 x) {
    unsigned short us = __builtin_bit_cast(unsigned short, x);
    unsigned u = ((unsigned)us) << 16;
    return __builtin_bit_cast(float, u);
}
__device__ __forceinline__ unsigned short f2bf_u(float f) {
    unsigned u = __builtin_bit_cast(unsigned, f);
    u += 0x7FFFu + ((u >> 16) & 1u);
    return (unsigned short)(u >> 16);
}
__device__ __forceinline__ __bf16 f2bf(float f) {
    return __builtin_bit_cast(__bf16, f2bf_u(f));
}
__device__ __forceinline__ unsigned pack2(float a, float b) {
    return (unsigned)f2bf_u(a) | ((unsigned)f2bf_u(b) << 16);
}

// ---------------------------------------------------------------------------
// WMMA fragment loads per CDNA5 ISA §7.12.2 layouts.
// A (16x32 bf16, MxK): lane m=lane&15, VGPR elems 0..7 = K[(hi*8)..+7],
//                      elems 8..15 = K[16+(hi*8)..+7], hi=lane>>4.
// B (32x16 bf16, KxN) from a ROW-MAJOR "W[N][K]" view (B[k][n]=W[n][k]):
//                      lane n=lane&15 loads K[(hi*8)..+7] and K[16+(hi*8)..+7], as A.
// C/D (16x16 f32): lane holds col N=lane&15, rows M = r + 8*hi for r=0..7.
// ---------------------------------------------------------------------------
__device__ __forceinline__ v16bf load_a_frag(const __bf16* base, int row_stride, int lane) {
    const __bf16* p = base + (size_t)(lane & 15) * row_stride + ((lane >> 4) << 3);
    union { v4u u[2]; v16bf v; } t;
    t.u[0] = *(const v4u*)(p);        // K +0..7   (16B)
    t.u[1] = *(const v4u*)(p + 16);   // K +16..23 (16B)
    return t.v;
}
__device__ __forceinline__ v16bf load_b_frag(const __bf16* base, int row_stride, int lane) {
    const __bf16* p = base + (size_t)(lane & 15) * row_stride + ((lane >> 4) << 3);
    union { v4u u[2]; v16bf v; } t;
    t.u[0] = *(const v4u*)(p);        // K +0..7
    t.u[1] = *(const v4u*)(p + 16);   // K +16..23
    return t.v;
}
__device__ __forceinline__ v8f wmma_bf16(v16bf a, v16bf b, v8f c) {
    return __builtin_amdgcn_wmma_f32_16x16x32_bf16(
        /*neg_a=*/false, a, /*neg_b=*/false, b,
        /*c_mod=*/(short)0, c, /*reuse_a=*/false, /*reuse_b=*/false);
}

// ---------------------------------------------------------------------------
// 1) fp32 -> bf16 conversion
// ---------------------------------------------------------------------------
__global__ __launch_bounds__(256) void cvt_bf16_kernel(const float* __restrict__ src,
                                                       __bf16* __restrict__ dst, int n) {
    int i = blockIdx.x * 256 + threadIdx.x;
    if (i < n) dst[i] = f2bf(src[i]);
}

// ---------------------------------------------------------------------------
// Shared GEMM core: C[32x64] per wave = A[32xK] @ W[64xK]^T via ping-pong
// double buffering (stage 0 computes while stage 1's loads fly, and vice
// versa). K must be a multiple of 64. No register copies in steady state.
// ---------------------------------------------------------------------------
template <int KDIM>
__device__ __forceinline__ void gemm_core_32x64(const __bf16* __restrict__ A, int m0,
                                                const __bf16* __restrict__ W, int n0,
                                                int lane, v8f acc[2][4]) {
    v16bf aF[2][2], bF[2][4];
    const __bf16* a0 = A + (size_t)m0 * KDIM;
    const __bf16* a1 = A + (size_t)(m0 + 16) * KDIM;

    // preload stage 0 (k = 0)
    aF[0][0] = load_a_frag(a0, KDIM, lane);
    aF[0][1] = load_a_frag(a1, KDIM, lane);
#pragma unroll
    for (int j = 0; j < 4; ++j)
        bF[0][j] = load_b_frag(W + (size_t)(n0 + j * 16) * KDIM, KDIM, lane);

    for (int k0 = 0; k0 < KDIM; k0 += 64) {
        // stage 1 loads (k0+32): always in range since KDIM % 64 == 0
        __builtin_prefetch(a0 + k0 + 64, 0, 1);
        aF[1][0] = load_a_frag(a0 + k0 + 32, KDIM, lane);
        aF[1][1] = load_a_frag(a1 + k0 + 32, KDIM, lane);
#pragma unroll
        for (int j = 0; j < 4; ++j)
            bF[1][j] = load_b_frag(W + (size_t)(n0 + j * 16) * KDIM + k0 + 32, KDIM, lane);

        // compute stage 0
#pragma unroll
        for (int j = 0; j < 4; ++j) {
            acc[0][j] = wmma_bf16(aF[0][0], bF[0][j], acc[0][j]);
            acc[1][j] = wmma_bf16(aF[0][1], bF[0][j], acc[1][j]);
        }

        // stage 0 loads for k0+64 (guarded; compiler peels the last trip)
        if (k0 + 64 < KDIM) {
            aF[0][0] = load_a_frag(a0 + k0 + 64, KDIM, lane);
            aF[0][1] = load_a_frag(a1 + k0 + 64, KDIM, lane);
#pragma unroll
            for (int j = 0; j < 4; ++j)
                bF[0][j] = load_b_frag(W + (size_t)(n0 + j * 16) * KDIM + k0 + 64, KDIM, lane);
        }

        // compute stage 1
#pragma unroll
        for (int j = 0; j < 4; ++j) {
            acc[0][j] = wmma_bf16(aF[1][0], bF[1][j], acc[0][j]);
            acc[1][j] = wmma_bf16(aF[1][1], bF[1][j], acc[1][j]);
        }
    }
}

// ---------------------------------------------------------------------------
// 2) QKV GEMM: qkv[4096,3072] = x_bf16[4096,1024] @ w_qkv[3072,1024]^T + bias
// ---------------------------------------------------------------------------
__global__ __launch_bounds__(256) void gemm_qkv_kernel(const __bf16* __restrict__ A,
                                                       const __bf16* __restrict__ W,
                                                       const float*  __restrict__ bias,
                                                       __bf16* __restrict__ out) {
    const int lane = threadIdx.x & 31;
    const int wave = threadIdx.x >> 5;
    const int m0 = blockIdx.x * 256 + wave * 32;
    const int n0 = blockIdx.y * 64;

    v8f acc[2][4] = {};
    gemm_core_32x64<CDIM>(A, m0, W, n0, lane, acc);

#pragma unroll
    for (int t = 0; t < 2; ++t) {
        const int mrow = m0 + t * 16 + ((lane >> 4) << 3);
#pragma unroll
        for (int j = 0; j < 4; ++j) {
            const int n = n0 + j * 16 + (lane & 15);
            const float bv = bias[n];
#pragma unroll
            for (int r = 0; r < 8; ++r)
                out[(size_t)(mrow + r) * QKVN + n] = f2bf(acc[t][j][r] + bv);
        }
    }
}

// ---------------------------------------------------------------------------
// 3) RoPE + scatter:  qkv[4096,3072] -> Q[B,H,L,D], K[B,H,L,D], V^T[B,H,D,L]
// ---------------------------------------------------------------------------
__global__ __launch_bounds__(256) void rope_scatter_kernel(const __bf16* __restrict__ qkv,
                                                           const float* __restrict__ cosT,
                                                           const float* __restrict__ sinT,
                                                           __bf16* __restrict__ qb,
                                                           __bf16* __restrict__ kb,
                                                           __bf16* __restrict__ vt) {
    int idx = blockIdx.x * 256 + threadIdx.x;     // over B*L*H*D = 2^22
    if (idx >= BATCH * SEQ * CDIM) return;
    const int d = idx & 63;
    const int h = (idx >> 6) & 15;
    const int l = (idx >> 10) & (SEQ - 1);
    const int b = idx >> 21;

    const __bf16* row = qkv + (size_t)(b * SEQ + l) * QKVN + h * HDIM;
    const float c = cosT[l * HDIM + d];
    const float s = sinT[l * HDIM + d];
    const int   d2  = (d < 32) ? d + 32 : d - 32;
    const float sgn = (d < 32) ? -1.f : 1.f;

    const float qv = bf2f(row[d]),          qp = bf2f(row[d2]);
    const float kv = bf2f(row[CDIM + d]),   kp = bf2f(row[CDIM + d2]);

    const size_t o = ((size_t)((b * HEADS + h) * SEQ + l)) * HDIM + d;
    qb[o] = f2bf(qv * c + sgn * qp * s);
    kb[o] = f2bf(kv * c + sgn * kp * s);
    // transposed V store: vt[b,h,d,l]
    vt[((size_t)(b * HEADS + h) * HDIM + d) * SEQ + l] = row[2 * CDIM + d];
}

// ---------------------------------------------------------------------------
// 4) Flash attention. 4 waves/block; each wave owns 16 q-rows.
//    S^T = K_tile(16x64) @ Q^T  (so softmax reduce is per-lane + shfl_xor 16)
//    out^T = V^T tiles @ P^T (P bounced through per-wave LDS [q][kv] tile)
// ---------------------------------------------------------------------------
__global__ __launch_bounds__(128) void attn_kernel(const __bf16* __restrict__ qb,
                                                   const __bf16* __restrict__ kb,
                                                   const __bf16* __restrict__ vt,
                                                   const int* __restrict__ mask,
                                                   __bf16* __restrict__ ctx) {
    __shared__ __bf16 lds[4][1024];
    const int lane = threadIdx.x & 31;
    const int wave = threadIdx.x >> 5;
    const int b = blockIdx.z, h = blockIdx.y;
    const int q0 = blockIdx.x * 64 + wave * 16;
    const int hi = lane >> 4;
    const int ql = lane & 15;

    const __bf16* Q = qb + (size_t)(b * HEADS + h) * SEQ * HDIM;
    const __bf16* K = kb + (size_t)(b * HEADS + h) * SEQ * HDIM;
    const __bf16* V = vt + (size_t)(b * HEADS + h) * HDIM * SEQ;
    const int*    M = mask + (size_t)b * SEQ * SEQ;

    // Q^T B-fragments (reused for the entire kv loop): d 0..31 and d 32..63
    const v16bf bq0 = load_b_frag(Q + (size_t)q0 * HDIM,      HDIM, lane);
    const v16bf bq1 = load_b_frag(Q + (size_t)q0 * HDIM + 32, HDIM, lane);

    v8f o0 = {}, o1 = {}, o2 = {}, o3 = {};
    float m_run = -1e30f, l_run = 0.f;
    const int qcol = q0 + ql;
    __bf16* P = &lds[wave][0];               // [16 q][32 kv] bf16 per wave

    for (int kv0 = 0; kv0 < SEQ; kv0 += 32) {
        // S^T tiles: rows = kv, cols = q. K=64 contraction -> 2 chained WMMAs.
        v8f s0 = {}, s1 = {};
        s0 = wmma_bf16(load_a_frag(K + (size_t)kv0 * HDIM,            HDIM, lane), bq0, s0);
        s0 = wmma_bf16(load_a_frag(K + (size_t)kv0 * HDIM + 32,       HDIM, lane), bq1, s0);
        s1 = wmma_bf16(load_a_frag(K + (size_t)(kv0 + 16) * HDIM,     HDIM, lane), bq0, s1);
        s1 = wmma_bf16(load_a_frag(K + (size_t)(kv0 + 16) * HDIM + 32,HDIM, lane), bq1, s1);

        // scale + mask + online softmax (column q lives in lane & lane^16)
        const int* mrowA = M + (size_t)qcol * SEQ + kv0 + hi * 8;
        const int* mrowB = mrowA + 16;
        float v[16];
        float tmax = -1e30f;
#pragma unroll
        for (int r = 0; r < 8; ++r) {
            float a = mrowA[r] ? s0[r] * 0.125f : -1e30f;
            float c = mrowB[r] ? s1[r] * 0.125f : -1e30f;
            v[r] = a; v[8 + r] = c;
            tmax = fmaxf(tmax, fmaxf(a, c));
        }
        tmax = fmaxf(tmax, __shfl_xor(tmax, 16));
        const float m_new = fmaxf(m_run, tmax);
        const float alpha = __expf(m_run - m_new);
        float lsum = 0.f;
#pragma unroll
        for (int r = 0; r < 16; ++r) { v[r] = __expf(v[r] - m_new); lsum += v[r]; }
        lsum += __shfl_xor(lsum, 16);
        l_run = l_run * alpha + lsum;
        m_run = m_new;
#pragma unroll
        for (int r = 0; r < 8; ++r) {
            o0[r] *= alpha; o1[r] *= alpha; o2[r] *= alpha; o3[r] *= alpha;
        }

        // P -> LDS as [q][kv] so the B reload is two 16-byte reads per lane
        {
            unsigned* pr = (unsigned*)(P + ql * 32 + hi * 8);
            v4u w0 = { pack2(v[0], v[1]),  pack2(v[2], v[3]),
                       pack2(v[4], v[5]),  pack2(v[6], v[7]) };
            v4u w1 = { pack2(v[8], v[9]),  pack2(v[10], v[11]),
                       pack2(v[12], v[13]), pack2(v[14], v[15]) };
            *(v4u*)pr        = w0;   // kv_local hi*8 .. +7
            *(v4u*)(pr + 8)  = w1;   // kv_local 16+hi*8 .. +7
        }
        const v16bf bp = load_b_frag(P, 32, lane);

        // out^T[d][q] += V^T rows (d) x P^T  -- 4 d-subtiles of 16
        o0 = wmma_bf16(load_a_frag(V + (size_t)(0 * 16) * SEQ + kv0, SEQ, lane), bp, o0);
        o1 = wmma_bf16(load_a_frag(V + (size_t)(1 * 16) * SEQ + kv0, SEQ, lane), bp, o1);
        o2 = wmma_bf16(load_a_frag(V + (size_t)(2 * 16) * SEQ + kv0, SEQ, lane), bp, o2);
        o3 = wmma_bf16(load_a_frag(V + (size_t)(3 * 16) * SEQ + kv0, SEQ, lane), bp, o3);
    }

    // normalize, transpose via LDS, write ctx[b, q, h*64+d]
    const float inv = 1.f / l_run;
    __bf16* Cb = &lds[wave][0];              // reuse as [16 q][64 d]
    {
        unsigned* cr = (unsigned*)(Cb + ql * 64 + hi * 8);
        v4u w0 = { pack2(o0[0]*inv, o0[1]*inv), pack2(o0[2]*inv, o0[3]*inv),
                   pack2(o0[4]*inv, o0[5]*inv), pack2(o0[6]*inv, o0[7]*inv) };
        v4u w1 = { pack2(o1[0]*inv, o1[1]*inv), pack2(o1[2]*inv, o1[3]*inv),
                   pack2(o1[4]*inv, o1[5]*inv), pack2(o1[6]*inv, o1[7]*inv) };
        v4u w2 = { pack2(o2[0]*inv, o2[1]*inv), pack2(o2[2]*inv, o2[3]*inv),
                   pack2(o2[4]*inv, o2[5]*inv), pack2(o2[6]*inv, o2[7]*inv) };
        v4u w3 = { pack2(o3[0]*inv, o3[1]*inv), pack2(o3[2]*inv, o3[3]*inv),
                   pack2(o3[4]*inv, o3[5]*inv), pack2(o3[6]*inv, o3[7]*inv) };
        *(v4u*)(cr)      = w0;   // d  0..15 half
        *(v4u*)(cr +  8) = w1;   // d 16..31 half
        *(v4u*)(cr + 16) = w2;   // d 32..47 half
        *(v4u*)(cr + 24) = w3;   // d 48..63 half
    }
    const unsigned* lu = (const unsigned*)Cb;
    unsigned* cu = (unsigned*)ctx;           // ctx rows: 512 dwords (1024 bf16)
    for (int i = lane; i < 512; i += 32) {
        const int qr = i >> 5, du = i & 31;
        cu[(size_t)(b * SEQ + q0 + qr) * (CDIM / 2) + h * (HDIM / 2) + du] = lu[i];
    }
}

// ---------------------------------------------------------------------------
// 5) Output projection: out_f32[4096,1024] = ctx_bf16 @ w_proj[1024,1024]^T
// ---------------------------------------------------------------------------
__global__ __launch_bounds__(256) void gemm_proj_kernel(const __bf16* __restrict__ A,
                                                        const __bf16* __restrict__ W,
                                                        float* __restrict__ out) {
    const int lane = threadIdx.x & 31;
    const int wave = threadIdx.x >> 5;
    const int m0 = blockIdx.x * 256 + wave * 32;
    const int n0 = blockIdx.y * 64;

    v8f acc[2][4] = {};
    gemm_core_32x64<CDIM>(A, m0, W, n0, lane, acc);

#pragma unroll
    for (int t = 0; t < 2; ++t) {
        const int mrow = m0 + t * 16 + ((lane >> 4) << 3);
#pragma unroll
        for (int j = 0; j < 4; ++j) {
            const int n = n0 + j * 16 + (lane & 15);
#pragma unroll
            for (int r = 0; r < 8; ++r)
                out[(size_t)(mrow + r) * CDIM + n] = acc[t][j][r];
        }
    }
}

// ---------------------------------------------------------------------------
// Host launcher
// ---------------------------------------------------------------------------
extern "C" void kernel_launch(void* const* d_in, const int* in_sizes, int n_in,
                              void* d_out, int out_size, void* d_ws, size_t ws_size,
                              hipStream_t stream) {
    (void)in_sizes; (void)n_in; (void)out_size; (void)ws_size;
    const float* x      = (const float*)d_in[0];   // [2,2048,1024]
    const float* w_qkv  = (const float*)d_in[1];   // [3072,1024]
    const float* b_qkv  = (const float*)d_in[2];   // [3072]
    const float* w_proj = (const float*)d_in[3];   // [1024,1024]
    const float* cosT   = (const float*)d_in[4];   // [2048,64]
    const float* sinT   = (const float*)d_in[5];   // [2048,64]
    const int*   mask   = (const int*)  d_in[6];   // [2,1,2048,2048]
    float* out = (float*)d_out;                    // [2,2048,1024] fp32

    // workspace carve-up (all 256B aligned)
    uint8_t* ws = (uint8_t*)d_ws;
    __bf16* xb   = (__bf16*)(ws);                       // 4096x1024      ( 8 MB)
    __bf16* wqb  = (__bf16*)(ws + 8388608);             // 3072x1024      ( 6 MB)
    __bf16* wpb  = (__bf16*)(ws + 14680064);            // 1024x1024      ( 2 MB)
    __bf16* qkvb = (__bf16*)(ws + 16777216);            // 4096x3072      (24 MB)
    __bf16* qb   = (__bf16*)(ws + 41943040);            // [B,H,L,D]      ( 8 MB)
    __bf16* kbuf = (__bf16*)(ws + 50331648);            // [B,H,L,D]      ( 8 MB)
    __bf16* vt   = (__bf16*)(ws + 58720256);            // [B,H,D,L]      ( 8 MB)
    __bf16* ctx  = (__bf16*)(ws + 67108864);            // 4096x1024      ( 8 MB)

    // 1) convert fp32 inputs to bf16
    cvt_bf16_kernel<<<(MROWS * CDIM + 255) / 256, 256, 0, stream>>>(x, xb, MROWS * CDIM);
    cvt_bf16_kernel<<<(QKVN * CDIM + 255) / 256, 256, 0, stream>>>(w_qkv, wqb, QKVN * CDIM);
    cvt_bf16_kernel<<<(CDIM * CDIM + 255) / 256, 256, 0, stream>>>(w_proj, wpb, CDIM * CDIM);

    // 2) QKV projection (+bias): grid (4096/256, 3072/64)
    gemm_qkv_kernel<<<dim3(MROWS / 256, QKVN / 64), 256, 0, stream>>>(xb, wqb, b_qkv, qkvb);

    // 3) RoPE + per-head scatter (V stored transposed)
    rope_scatter_kernel<<<(BATCH * SEQ * CDIM + 255) / 256, 256, 0, stream>>>(
        qkvb, cosT, sinT, qb, kbuf, vt);

    // 4) flash attention
    attn_kernel<<<dim3(SEQ / 64, HEADS, BATCH), 128, 0, stream>>>(qb, kbuf, vt, mask, ctx);

    // 5) output projection -> fp32: grid (4096/256, 1024/64)
    gemm_proj_kernel<<<dim3(MROWS / 256, CDIM / 64), 256, 0, stream>>>(ctx, wpb, out);
}
